// PointnetSAModuleMSG_37684043055437
// MI455X (gfx1250) — matrix operand order repaired and checked by reading, not compiled, not run on hardware
//
#include <hip/hip_runtime.h>

typedef __attribute__((ext_vector_type(16))) _Float16 v16h;
typedef __attribute__((ext_vector_type(8)))  _Float16 v8h;
typedef __attribute__((ext_vector_type(8)))  float    v8f;

union V16 { v16h v; struct { v8h lo; v8h hi; } p; };

#define NPTS   8192
#define NB     4
#define NPOINT 2048
#define NFEAT  16

// ---------------------------------------------------------------------------
// Kernel 1: Farthest Point Sampling. One workgroup (1024 thr = 32 waves) per
// batch. Point coords + running min-dist live in registers (8 pts/thread).
// Argmax via wave32 shfl_xor butterfly + 32-slot LDS stage (2 barriers/iter).
// Key packs (distBits<<32)|(~idx) so max-key == max dist, ties -> lowest idx,
// matching jnp.argmax semantics (dists are >= 0 so f32 bits are monotonic).
// ---------------------------------------------------------------------------
__global__ __launch_bounds__(1024) void fps_kernel(const float* __restrict__ xyz,
                                                   float* __restrict__ new_xyz) {
  const int b = blockIdx.x;
  const int t = threadIdx.x;
  const float* X = xyz + (size_t)b * NPTS * 3;
  float px[8], py[8], pz[8], dd[8];
#pragma unroll
  for (int k = 0; k < 8; ++k) {
    int j = t + 1024 * k;
    px[k] = X[j * 3 + 0]; py[k] = X[j * 3 + 1]; pz[k] = X[j * 3 + 2];
    dd[k] = 1e10f;
  }
  __shared__ unsigned long long red[32];
  __shared__ int winner;
  float lx = X[0], ly = X[1], lz = X[2];
  if (t == 0) {
    float* o = new_xyz + (size_t)b * NPOINT * 3;
    o[0] = lx; o[1] = ly; o[2] = lz;
  }
  const int lane = t & 31, wv = t >> 5;
  for (int it = 1; it < NPOINT; ++it) {
    float bestd = -1.0f; int bestj = 0;
#pragma unroll
    for (int k = 0; k < 8; ++k) {
      float dx = px[k] - lx, dy = py[k] - ly, dz = pz[k] - lz;
      float d = dx * dx + dy * dy + dz * dz;
      dd[k] = fminf(dd[k], d);
      if (dd[k] > bestd) { bestd = dd[k]; bestj = t + 1024 * k; }
    }
    unsigned long long key =
        ((unsigned long long)__float_as_uint(bestd) << 32) |
        (unsigned)(0xFFFFFFFFu - (unsigned)bestj);
#pragma unroll
    for (int m = 16; m >= 1; m >>= 1) {
      unsigned hi = (unsigned)(key >> 32), lo = (unsigned)key;
      unsigned hi2 = __shfl_xor(hi, m, 32);
      unsigned lo2 = __shfl_xor(lo, m, 32);
      unsigned long long o = ((unsigned long long)hi2 << 32) | lo2;
      if (o > key) key = o;
    }
    if (lane == 0) red[wv] = key;
    __syncthreads();
    if (wv == 0) {
      unsigned long long k2 = red[lane];
#pragma unroll
      for (int m = 16; m >= 1; m >>= 1) {
        unsigned hi = (unsigned)(k2 >> 32), lo = (unsigned)k2;
        unsigned hi2 = __shfl_xor(hi, m, 32);
        unsigned lo2 = __shfl_xor(lo, m, 32);
        unsigned long long o = ((unsigned long long)hi2 << 32) | lo2;
        if (o > k2) k2 = o;
      }
      if (lane == 0) winner = (int)(0xFFFFFFFFu - (unsigned)k2);
    }
    __syncthreads();
    const int w = winner;
    lx = X[w * 3 + 0]; ly = X[w * 3 + 1]; lz = X[w * 3 + 2];
    if (t == 0) {
      float* o = new_xyz + ((size_t)b * NPOINT + it) * 3;
      o[0] = lx; o[1] = ly; o[2] = lz;
    }
  }
}

// ---------------------------------------------------------------------------
// Kernel 2: ball query, both radii in one scan (r0 subset of r1). One thread
// per (b,s); all threads of a block share batch b and scan the same j stream
// (broadcast-friendly, xyz fully L2-resident). Indices land in scan order ==
// ascending order == reference top_k(-index) order.
// ---------------------------------------------------------------------------
__global__ __launch_bounds__(256) void ballq_kernel(const float* __restrict__ xyz,
                                                    const float* __restrict__ nxyz,
                                                    int* __restrict__ idx0,
                                                    int* __restrict__ idx1) {
  const int gid = blockIdx.x * 256 + threadIdx.x;   // b*2048 + s
  const int b = gid >> 11;
  const float* X = xyz + (size_t)b * NPTS * 3;
  const float nx = nxyz[gid * 3 + 0], ny = nxyz[gid * 3 + 1], nz = nxyz[gid * 3 + 2];
  int* o0 = idx0 + (size_t)gid * 16;
  int* o1 = idx1 + (size_t)gid * 32;
  const float r0sq = 0.2f * 0.2f, r1sq = 0.4f * 0.4f;
  int c0 = 0, c1 = 0, f0 = 0, f1 = 0;
  for (int j = 0; j < NPTS; ++j) {
    float dx = X[j * 3 + 0] - nx;
    float dy = X[j * 3 + 1] - ny;
    float dz = X[j * 3 + 2] - nz;
    float d2 = dx * dx + dy * dy + dz * dz;
    if (d2 < r1sq) {
      if (c1 < 32) { if (c1 == 0) f1 = j; o1[c1++] = j; }
      if (d2 < r0sq && c0 < 16) { if (c0 == 0) f0 = j; o0[c0++] = j; }
      if (c1 == 32 && c0 == 16) break;
    }
  }
  for (int k = c0; k < 16; ++k) o0[k] = f0;
  for (int k = c1; k < 32; ++k) o1[k] = f1;
}

// ---------------------------------------------------------------------------
// Fused group + shared-MLP + max-pool using v_wmma_f32_16x16x32_f16.
// ---------------------------------------------------------------------------
struct LayerP { const float *w, *g, *b, *m, *v; };
struct MlpP { LayerP l[3]; };

// Pre-swizzle weights (BN scale folded) into LDS in exact B-matrix lane
// layout: tile (kt,nt) -> 32 lanes x 16 contiguous halves; lane ln half h is
// element K = kt*32 + (ln>>4)*8 + (h&7) + (h>>3)*16, N = nt*16 + (ln&15).
__device__ inline void fill_w(_Float16* dst, const float* w, const float* gamma,
                              const float* var, int Cin, int Cout, int Kpad,
                              int tid, int nthr) {
  const int NT = Cout / 16;
  const int total = Kpad * Cout;
  for (int i = tid; i < total; i += nthr) {
    int tile = i >> 9, rem = i & 511;
    int ln = rem >> 4, h = rem & 15;
    int kt = tile / NT, nt = tile % NT;
    int k = kt * 32 + (ln >> 4) * 8 + (h & 7) + (h >> 3) * 16;
    int n = nt * 16 + (ln & 15);
    float a = gamma[n] * rsqrtf(var[n] + 1e-5f);
    dst[i] = (_Float16)((k < Cin) ? w[n * Cin + k] * a : 0.0f);
  }
}

// One MLP layer: A from row-major f16 LDS (ISA 16-bit A layout = two 16B
// chunks per lane), B from pre-swizzled LDS, C init = folded BN bias, ReLU,
// either restage to LDS (row-major for next layer) or fold into max-pool acc.
template <int MT, int KT, int NT, int KIN, int KOUT, bool LAST>
__device__ inline void run_layer(const _Float16* actIn, _Float16* actOut,
                                 const _Float16* wl, const float* bias,
                                 int lane, v8f* macc) {
  const int lh = lane & 15, lq = lane >> 4;
#pragma unroll
  for (int mt = 0; mt < MT; ++mt) {
    V16 a[KT];
#pragma unroll
    for (int kt = 0; kt < KT; ++kt) {
      const _Float16* base = actIn + (mt * 16 + lh) * KIN + kt * 32 + lq * 8;
      a[kt].p.lo = *(const v8h*)base;        // lanes 0-15: K 0-7 | 16-31: K 8-15
      a[kt].p.hi = *(const v8h*)(base + 16); // lanes 0-15: K16-23 | 16-31: K24-31
    }
#pragma unroll
    for (int nt = 0; nt < NT; ++nt) {
      float bv = bias[nt * 16 + lh];
      v8f c = {bv, bv, bv, bv, bv, bv, bv, bv};
#pragma unroll
      for (int kt = 0; kt < KT; ++kt) {
        v16h bm = *(const v16h*)(wl + ((size_t)(kt * NT + nt) * 32 + lane) * 16);
        c = __builtin_amdgcn_wmma_f32_16x16x32_f16(false, a[kt].v, false, bm,
                                                   (short)0, c, false, false);
      }
#pragma unroll
      for (int r = 0; r < 8; ++r) c[r] = fmaxf(c[r], 0.0f);
      if constexpr (LAST) {
#pragma unroll
        for (int r = 0; r < 8; ++r) macc[nt][r] = fmaxf(macc[nt][r], c[r]);
      } else {
        _Float16* orow = actOut + (mt * 16 + lq * 8) * KOUT + nt * 16 + lh;
#pragma unroll
        for (int r = 0; r < 8; ++r) orow[r * KOUT] = (_Float16)c[r];
      }
    }
  }
}

template <int NS, int C1, int C2, int C3, int GPW, int WAVES>
__global__ __launch_bounds__(256) void mlp_kernel(const float* __restrict__ xyz,
                                                  const float* __restrict__ feat,
                                                  const int* __restrict__ idx,
                                                  const float* __restrict__ nxyz,
                                                  float* __restrict__ outF,
                                                  int chanBase, MlpP P) {
  constexpr int K0 = 32, K1 = C1, K2 = C2;
  constexpr int MT = NS / 16;
  constexpr int KT1 = K0 / 32, KT2 = K1 / 32, KT3 = K2 / 32;
  constexpr int NT1 = C1 / 16, NT2 = C2 / 16, NT3 = C3 / 16;
  constexpr int MAXK = (K1 > K2) ? K1 : K2;

  extern __shared__ char smem[];
  _Float16* wl1 = (_Float16*)smem;
  _Float16* wl2 = wl1 + K0 * C1;
  _Float16* wl3 = wl2 + K1 * C2;
  float* bias = (float*)(wl3 + K2 * C3);
  _Float16* act = (_Float16*)(bias + C1 + C2 + C3);

  fill_w(wl1, P.l[0].w, P.l[0].g, P.l[0].v, 19, C1, K0, threadIdx.x, blockDim.x);
  fill_w(wl2, P.l[1].w, P.l[1].g, P.l[1].v, C1, C2, K1, threadIdx.x, blockDim.x);
  fill_w(wl3, P.l[2].w, P.l[2].g, P.l[2].v, C2, C3, K2, threadIdx.x, blockDim.x);
  for (int i = threadIdx.x; i < C1; i += blockDim.x) {
    float a = P.l[0].g[i] * rsqrtf(P.l[0].v[i] + 1e-5f);
    bias[i] = P.l[0].b[i] - P.l[0].m[i] * a;
  }
  for (int i = threadIdx.x; i < C2; i += blockDim.x) {
    float a = P.l[1].g[i] * rsqrtf(P.l[1].v[i] + 1e-5f);
    bias[C1 + i] = P.l[1].b[i] - P.l[1].m[i] * a;
  }
  for (int i = threadIdx.x; i < C3; i += blockDim.x) {
    float a = P.l[2].g[i] * rsqrtf(P.l[2].v[i] + 1e-5f);
    bias[C1 + C2 + i] = P.l[2].b[i] - P.l[2].m[i] * a;
  }
  __syncthreads();

  const int lane = threadIdx.x & 31, wv = threadIdx.x >> 5;
  _Float16* aA = act + (size_t)wv * (2 * NS * MAXK);
  _Float16* aB = aA + NS * MAXK;

#pragma unroll 1
  for (int g = 0; g < GPW; ++g) {
    const int sg = (blockIdx.x * WAVES + wv) * GPW + g;
    const int b = sg >> 11, s = sg & 2047;
    const int* gi = idx + (size_t)sg * NS;
    if (g + 1 < GPW) __builtin_prefetch(gi + NS, 0, 1);
    const float nx = nxyz[sg * 3 + 0], ny = nxyz[sg * 3 + 1], nz = nxyz[sg * 3 + 2];
    if (lane < NS) {  // gather group rows into A0 (row-major, Kpad=32)
      int j = gi[lane];
      const float* p = xyz + ((size_t)b * NPTS + j) * 3;
      _Float16* row = aA + lane * K0;
      row[0] = (_Float16)(p[0] - nx);
      row[1] = (_Float16)(p[1] - ny);
      row[2] = (_Float16)(p[2] - nz);
      const float* fp = feat + (size_t)b * NFEAT * NPTS + j;
#pragma unroll
      for (int c = 0; c < NFEAT; ++c) row[3 + c] = (_Float16)fp[(size_t)c * NPTS];
#pragma unroll
      for (int k = 3 + NFEAT; k < K0; ++k) row[k] = (_Float16)0.0f;
    }
    v8f macc[NT3];
#pragma unroll
    for (int nt = 0; nt < NT3; ++nt)
      macc[nt] = (v8f){0.f, 0.f, 0.f, 0.f, 0.f, 0.f, 0.f, 0.f};

    run_layer<MT, KT1, NT1, K0, K1, false>(aA, aB, wl1, bias, lane, nullptr);
    run_layer<MT, KT2, NT2, K1, K2, false>(aB, aA, wl2, bias + C1, lane, nullptr);
    run_layer<MT, KT3, NT3, K2, K2, true>(aA, nullptr, wl3, bias + C1 + C2, lane, macc);

#pragma unroll
    for (int nt = 0; nt < NT3; ++nt) {
      float pm = macc[nt][0];
#pragma unroll
      for (int r = 1; r < 8; ++r) pm = fmaxf(pm, macc[nt][r]);
      pm = fmaxf(pm, __shfl_xor(pm, 16, 32));   // merge M halves (lanes n / n+16)
      if (lane < 16)
        outF[((size_t)b * 192 + chanBase + nt * 16 + lane) * NPOINT + s] = pm;
    }
  }
}

// ---------------------------------------------------------------------------
extern "C" void kernel_launch(void* const* d_in, const int* in_sizes, int n_in,
                              void* d_out, int out_size, void* d_ws, size_t ws_size,
                              hipStream_t stream) {
  (void)in_sizes; (void)n_in; (void)out_size; (void)ws_size;
  const float* xyz  = (const float*)d_in[0];
  const float* feat = (const float*)d_in[1];
  MlpP p0, p1;
  for (int i = 0; i < 3; ++i) {
    p0.l[i].w = (const float*)d_in[2 + i * 5 + 0];
    p0.l[i].g = (const float*)d_in[2 + i * 5 + 1];
    p0.l[i].b = (const float*)d_in[2 + i * 5 + 2];
    p0.l[i].m = (const float*)d_in[2 + i * 5 + 3];
    p0.l[i].v = (const float*)d_in[2 + i * 5 + 4];
    p1.l[i].w = (const float*)d_in[17 + i * 5 + 0];
    p1.l[i].g = (const float*)d_in[17 + i * 5 + 1];
    p1.l[i].b = (const float*)d_in[17 + i * 5 + 2];
    p1.l[i].m = (const float*)d_in[17 + i * 5 + 3];
    p1.l[i].v = (const float*)d_in[17 + i * 5 + 4];
  }
  float* out_nxyz = (float*)d_out;                         // (4,2048,3)
  float* out_feat = out_nxyz + (size_t)NB * NPOINT * 3;    // (4,192,2048)
  int* idx0 = (int*)d_ws;                                  // (4,2048,16)
  int* idx1 = idx0 + (size_t)NB * NPOINT * 16;             // (4,2048,32)

  fps_kernel<<<NB, 1024, 0, stream>>>(xyz, out_nxyz);
  ballq_kernel<<<NB * (NPOINT / 256), 256, 0, stream>>>(xyz, out_nxyz, idx0, idx1);

  // scale 0: 19->32->32->64, ns=16; 8 waves/block, 8 groups/wave -> 128 blocks
  {
    constexpr size_t lds = (32 * 32 + 32 * 32 + 32 * 64) * 2  // weights
                         + (32 + 32 + 64) * 4                 // bias
                         + 8ull * 2 * 16 * 32 * 2;            // act ping-pong
    mlp_kernel<16, 32, 32, 64, 8, 8><<<128, 256, lds, stream>>>(
        xyz, feat, idx0, out_nxyz, out_feat, 0, p0);
  }
  // scale 1: 19->64->64->128, ns=32; 4 waves/block, 8 groups/wave -> 256 blocks
  {
    constexpr size_t lds = (32 * 64 + 64 * 64 + 64 * 128) * 2
                         + (64 + 64 + 128) * 4
                         + 4ull * 2 * 32 * 64 * 2;
    mlp_kernel<32, 64, 64, 128, 8, 4><<<256, 128, lds, stream>>>(
        xyz, feat, idx1, out_nxyz, out_feat, 64, p1);
  }
}
